// multihead_attentive_layer_86801289052505
// MI455X (gfx1250) — matrix operand order repaired and checked by reading, not compiled, not run on hardware
//
#include <hip/hip_runtime.h>
#include <hip/hip_bf16.h>

typedef __attribute__((ext_vector_type(16))) _Float16 v16h;
typedef __attribute__((ext_vector_type(8)))  _Float16 v8h;
typedef __attribute__((ext_vector_type(8)))  float    v8f;

#define WMMA_F16(a, b, c) __builtin_amdgcn_wmma_f32_16x16x32_f16( \
    false, (a), false, (b), (short)0, (c), false, false)

__device__ __forceinline__ v16h cat8(v8h lo, v8h hi) {
  return __builtin_shufflevector(lo, hi, 0,1,2,3,4,5,6,7,8,9,10,11,12,13,14,15);
}

// ---- explicit global (addrspace(1)) accessors: force global_load/global_store.
#define GAS __attribute__((address_space(1)))
__device__ __forceinline__ v8h gload8(const _Float16* p) {
  return *(const GAS v8h*)(unsigned long long)p;
}
__device__ __forceinline__ float gload_f(const float* p) {
  return *(const GAS float*)(unsigned long long)p;
}
__device__ __forceinline__ void gstore_h(_Float16* p, _Float16 v) {
  *(GAS _Float16*)(unsigned long long)p = v;
}
__device__ __forceinline__ void gstore_f(float* p, float v) {
  *(GAS float*)(unsigned long long)p = v;
}
__device__ __forceinline__ void gstore8(_Float16* p, v8h v) {
  *(GAS v8h*)(unsigned long long)p = v;
}

// ---- CDNA5 async global->LDS copy, GVS mode: SGPR64 base + VGPR32 offset.
// Base is LOOP-INVARIANT (stays in SGPRs); the K-step advance lives in a
// running 32-bit VGPR offset. INST_OFFSET is added to BOTH the global and
// LDS addresses (ISA 08 §4.4), so the +16B A-chunk reuses the same operands.
__device__ __forceinline__ void async_gvs(unsigned lds, unsigned voff,
                                          unsigned long long sbase) {
  asm volatile("global_load_async_to_lds_b128 %0, %1, %2"
               :: "v"(lds), "v"(voff), "s"(sbase) : "memory");
}
__device__ __forceinline__ void async_gvs16(unsigned lds, unsigned voff,
                                            unsigned long long sbase) {
  asm volatile("global_load_async_to_lds_b128 %0, %1, %2 offset:16"
               :: "v"(lds), "v"(voff), "s"(sbase) : "memory");
}
#define WAIT_ASYNC0() asm volatile("s_wait_asynccnt 0x0" ::: "memory")
#define WAIT_DS0()    asm volatile("s_wait_dscnt 0x0" ::: "memory")

// ---------------- problem constants ----------------
#define BATCH   2
#define SEQ     2048
#define DMODEL  1024
#define HEADS   16
#define HDIM    128
#define ATT     2048   // HEADS*HDIM
#define OUTU    1024
#define ROWS    (BATCH*SEQ)   // 4096

// ---------------- workspace layout (bytes) ----------------
#define OFF_XH   ((size_t)0)            //  8 MB  f16 X        [4096][1024]
#define OFF_WQT  ((size_t)8388608)      //  4 MB  f16 Wq^T     [2048][1024]
#define OFF_WKT  ((size_t)12582912)     //  4 MB
#define OFF_WVT  ((size_t)16777216)     //  4 MB
#define OFF_WOT  ((size_t)20971520)     //  4 MB  f16 Wo^T     [1024][2048]
#define OFF_Q    ((size_t)25165824)     // 16 MB  f16 Q  [B][H][S][dh]
#define OFF_K    ((size_t)41943040)     // 16 MB  f16 K  [B][H][S][dh]
#define OFF_VT   ((size_t)58720256)     // 16 MB  f16 Vt [B][H][dh][S]
#define OFF_CTX  ((size_t)75497472)     // 16 MB  f16 ctx [4096][2048]

// ---------------- prep kernels ----------------
__global__ void cvt_f32_f16(const float* __restrict__ src,
                            _Float16* __restrict__ dst, int n) {
  int i = blockIdx.x * 256 + threadIdx.x;
  if (i < n) gstore_h(dst + i, (_Float16)gload_f(src + i));
}

// src [K][N] f32 row-major -> dst [N][K] f16 row-major
__global__ void transpose_w(const float* __restrict__ src,
                            _Float16* __restrict__ dst, int K, int N) {
  int i = blockIdx.x * 256 + threadIdx.x;
  if (i >= K * N) return;
  int n = i / K, k = i % K;
  gstore_h(dst + (size_t)n * K + k, (_Float16)gload_f(src + (size_t)k * N + n));
}

// ---------------- generic WMMA GEMM ----------------
// C[M][N] = A[M][K] * B[K][N] + bias[N], with B supplied transposed (Bt[N][K]).
// Block tile 128x64, 8 waves, each wave 32x32 (2x2 WMMA tiles), K step 32.
// Double-buffered GLOBAL_LOAD_ASYNC_TO_LDS staging (ASYNCcnt), GVS addressing.
// K loop held at unroll 2 so the double-buffer folds but offsets stay a
// single loop-carried register (no fully-unrolled address precompute).
// MODE 1: f16 out, split-head store [B][H][S][dh]
// MODE 2: f16 out, transposed split-head store [B][H][dh][S] (packed b128)
// MODE 3: f32 out, row-major [M][1024]
#define LSTR 40

template<int MODE>
__device__ void gemm_body(const _Float16* __restrict__ A,
                          const _Float16* __restrict__ Bt,
                          const float* __restrict__ bias,
                          void* __restrict__ Out, int Ktot) {
  __shared__ _Float16 sA[2][128 * LSTR];
  __shared__ _Float16 sB[2][64 * LSTR];
  const int tid   = threadIdx.x;
  const int lane  = tid & 31;
  const int wave  = tid >> 5;
  const int wm    = wave & 3;     // wave row group 0..3
  const int wn    = wave >> 2;    // wave col group 0..1
  const int mBase = blockIdx.y * 128;
  const int nBase = blockIdx.x * 64;
  const int mlane = lane & 15;
  const int hi    = lane >> 4;

  const v8f zero8 = {0.f,0.f,0.f,0.f,0.f,0.f,0.f,0.f};
  v8f acc[2][2];
  #pragma unroll
  for (int a = 0; a < 2; ++a)
    #pragma unroll
    for (int b = 0; b < 2; ++b) acc[a][b] = zero8;

  const int arow = tid >> 1;          // 0..127
  const int aoff = (tid & 1) * 16;    // 0,16
  const int brow = tid >> 2;          // 0..63
  const int boff = (tid & 3) * 8;     // 0,8,16,24

  // async addressing: loop-invariant SGPR bases, running 32-bit VGPR offsets
  const unsigned long long baseA = (unsigned long long)(A  + (size_t)mBase * Ktot);
  const unsigned long long baseB = (unsigned long long)(Bt + (size_t)nBase * Ktot);
  const unsigned ldsA0 = (unsigned)(unsigned long long)&sA[0][arow * LSTR + aoff];
  const unsigned ldsA1 = (unsigned)(unsigned long long)&sA[1][arow * LSTR + aoff];
  const unsigned ldsB0 = (unsigned)(unsigned long long)&sB[0][brow * LSTR + boff];
  const unsigned ldsB1 = (unsigned)(unsigned long long)&sB[1][brow * LSTR + boff];
  unsigned vA = (unsigned)((arow * Ktot + aoff) * 2);   // bytes, +64/tile
  unsigned vB = (unsigned)((brow * Ktot + boff) * 2);   // bytes, +64/tile

  // stage tile 0 into buffer 0
  async_gvs  (ldsA0, vA, baseA);
  async_gvs16(ldsA0, vA, baseA);
  async_gvs  (ldsB0, vB, baseB);

  int pb = 0;
  #pragma clang loop unroll_count(2)
  for (int k0 = 0; k0 < Ktot; k0 += 32) {
    WAIT_ASYNC0();        // own async copies into buf pb are done
    __syncthreads();      // everyone's copies done / previous compute done
    if (k0 + 32 < Ktot) {
      vA += 64; vB += 64;
      const unsigned la = pb ? ldsA0 : ldsA1;   // next buffer = pb^1
      const unsigned lb = pb ? ldsB0 : ldsB1;
      async_gvs  (la, vA, baseA);
      async_gvs16(la, vA, baseA);
      async_gvs  (lb, vB, baseB);
    }

    v16h afrag[2], bfrag[2];
    #pragma unroll
    for (int mt = 0; mt < 2; ++mt) {
      const _Float16* p = &sA[pb][(wm * 32 + mt * 16 + mlane) * LSTR + hi * 8];
      afrag[mt] = cat8(*(const v8h*)p, *(const v8h*)(p + 16));
    }
    #pragma unroll
    for (int nt = 0; nt < 2; ++nt) {
      const _Float16* p = &sB[pb][(wn * 32 + nt * 16 + mlane) * LSTR + hi * 8];
      bfrag[nt] = cat8(*(const v8h*)p, *(const v8h*)(p + 16));
    }
    #pragma unroll
    for (int mt = 0; mt < 2; ++mt)
      #pragma unroll
      for (int nt = 0; nt < 2; ++nt)
        acc[mt][nt] = WMMA_F16(afrag[mt], bfrag[nt], acc[mt][nt]);
    pb ^= 1;
  }

  #pragma unroll
  for (int mt = 0; mt < 2; ++mt) {
    #pragma unroll
    for (int nt = 0; nt < 2; ++nt) {
      const int gn = nBase + wn * 32 + nt * 16 + mlane;
      const float bval = gload_f(bias + gn);
      if constexpr (MODE == 2) {
        // fixed d column, 8 contiguous s rows per lane -> single b128 store
        const int gm0 = mBase + wm * 32 + mt * 16 + hi * 8;
        const int b = gm0 >> 11, s0 = gm0 & (SEQ - 1);
        const int h = gn >> 7,  d = gn & (HDIM - 1);
        v8h pk;
        #pragma unroll
        for (int i = 0; i < 8; ++i) pk[i] = (_Float16)(acc[mt][nt][i] + bval);
        gstore8((_Float16*)Out + ((size_t)(b * HEADS + h) * HDIM + d) * SEQ + s0, pk);
      } else {
        #pragma unroll
        for (int i = 0; i < 8; ++i) {
          const int gm = mBase + wm * 32 + mt * 16 + i + hi * 8;
          float v = acc[mt][nt][i] + bval;
          if constexpr (MODE == 3) {
            gstore_f((float*)Out + (size_t)gm * OUTU + gn, v);
          } else {
            const int b = gm >> 11, s = gm & (SEQ - 1);
            const int h = gn >> 7,  d = gn & (HDIM - 1);
            gstore_h((_Float16*)Out + ((size_t)(b * HEADS + h) * SEQ + s) * HDIM + d,
                     (_Float16)v);
          }
        }
      }
    }
  }
}

__global__ void __launch_bounds__(256)
gemm_qkv_kernel(const _Float16* __restrict__ X,
                const _Float16* __restrict__ WQT, const _Float16* __restrict__ WKT,
                const _Float16* __restrict__ WVT,
                const float* __restrict__ bq, const float* __restrict__ bk,
                const float* __restrict__ bv,
                _Float16* __restrict__ Q, _Float16* __restrict__ Ko,
                _Float16* __restrict__ VT) {
  const int z = blockIdx.z;
  if (z == 0)      gemm_body<1>(X, WQT, bq, Q,  DMODEL);
  else if (z == 1) gemm_body<1>(X, WKT, bk, Ko, DMODEL);
  else             gemm_body<2>(X, WVT, bv, VT, DMODEL);
}

__global__ void __launch_bounds__(256)
gemm_out_kernel(const _Float16* __restrict__ CTX, const _Float16* __restrict__ WOT,
                const float* __restrict__ bo, float* __restrict__ Out) {
  gemm_body<3>(CTX, WOT, bo, Out, ATT);
}

// ---------------- flash attention ----------------
// One wave per 16-row q-tile. Streams 32 keys / iteration:
//   scores: 8 WMMAs (2 key sub-tiles x 4 dh chunks), online softmax,
//   P via LDS (C-layout -> A-layout, s_wait_dscnt), ctx: 8 WMMAs.
#define OSTR 136   // sOut row stride (halves): 272B keeps 16B alignment, spreads banks

__global__ void __launch_bounds__(128)
attn_kernel(const _Float16* __restrict__ Q, const _Float16* __restrict__ Km,
            const _Float16* __restrict__ VT, _Float16* __restrict__ CTX) {
  __shared__ _Float16 sP[4][16 * LSTR];
  __shared__ _Float16 sOut[4][16 * OSTR];
  const int lane  = threadIdx.x & 31;
  const int wave  = threadIdx.x >> 5;
  const int mlane = lane & 15;
  const int hi    = lane >> 4;
  const int bh    = blockIdx.y;              // b*HEADS + h
  const int qbase = (blockIdx.x * 4 + wave) * 16;
  const int b     = bh >> 4, h = bh & 15;
  const float rscale = 0.022097086912079608f;  // 1/sqrt(2048)

  // Q fragments kept resident (16 x 128, 4 K-chunks)
  v16h qa[4];
  {
    const _Float16* qrow = Q + ((size_t)bh * SEQ + qbase + mlane) * HDIM;
    #pragma unroll
    for (int c = 0; c < 4; ++c) {
      const _Float16* p = qrow + c * 32 + hi * 8;
      qa[c] = cat8(gload8(p), gload8(p + 16));
    }
  }

  const v8f zero8 = {0.f,0.f,0.f,0.f,0.f,0.f,0.f,0.f};
  v8f o[8];
  #pragma unroll
  for (int nt = 0; nt < 8; ++nt) o[nt] = zero8;
  float mrow[8], lrow[8];
  #pragma unroll
  for (int i = 0; i < 8; ++i) { mrow[i] = -1.0e30f; lrow[i] = 0.f; }

  _Float16* myP = &sP[wave][0];

  for (int kb = 0; kb < qbase + 16; kb += 32) {
    v8f s0 = zero8, s1 = zero8;
    #pragma unroll
    for (int c = 0; c < 4; ++c) {
      const _Float16* p0 = Km + ((size_t)bh * SEQ + kb + mlane) * HDIM + c * 32 + hi * 8;
      const _Float16* p1 = p0 + 16 * HDIM;
      v16h b0 = cat8(gload8(p0), gload8(p0 + 16));
      v16h b1 = cat8(gload8(p1), gload8(p1 + 16));
      s0 = WMMA_F16(qa[c], b0, s0);
      s1 = WMMA_F16(qa[c], b1, s1);
    }

    // scale + causal mask + row stats (lanes 0-15 rows i, lanes 16-31 rows i+8)
    float corr[8];
    #pragma unroll
    for (int i = 0; i < 8; ++i) {
      const int q = qbase + i + hi * 8;
      float a  = s0[i] * rscale;
      float c2 = s1[i] * rscale;
      if (kb + mlane > q)      a  = -1.0e30f;
      if (kb + 16 + mlane > q) c2 = -1.0e30f;
      float t = fmaxf(a, c2);
      t = fmaxf(t, __shfl_xor(t, 1));
      t = fmaxf(t, __shfl_xor(t, 2));
      t = fmaxf(t, __shfl_xor(t, 4));
      t = fmaxf(t, __shfl_xor(t, 8));
      const float mn = fmaxf(mrow[i], t);
      corr[i] = __expf(mrow[i] - mn);
      mrow[i] = mn;
      a  = __expf(a  - mn);
      c2 = __expf(c2 - mn);
      s0[i] = a; s1[i] = c2;
      float rs = a + c2;
      rs += __shfl_xor(rs, 1);
      rs += __shfl_xor(rs, 2);
      rs += __shfl_xor(rs, 4);
      rs += __shfl_xor(rs, 8);
      lrow[i] = lrow[i] * corr[i] + rs;
    }
    #pragma unroll
    for (int nt = 0; nt < 8; ++nt)
      #pragma unroll
      for (int i = 0; i < 8; ++i) o[nt][i] *= corr[i];

    // P: C-layout -> row-major [16][32] in LDS -> A-fragment
    #pragma unroll
    for (int i = 0; i < 8; ++i) {
      const int r = i + hi * 8;
      myP[r * LSTR + mlane]      = (_Float16)s0[i];
      myP[r * LSTR + 16 + mlane] = (_Float16)s1[i];
    }
    WAIT_DS0();
    const _Float16* pp = myP + mlane * LSTR + hi * 8;
    v16h pf = cat8(*(const v8h*)pp, *(const v8h*)(pp + 16));

    // ctx += P @ V  (B from transposed V: contiguous keys)
    #pragma unroll
    for (int nt = 0; nt < 8; ++nt) {
      const _Float16* vp = VT + ((size_t)bh * HDIM + nt * 16 + mlane) * SEQ + kb + hi * 8;
      v16h vf = cat8(gload8(vp), gload8(vp + 16));
      o[nt] = WMMA_F16(pf, vf, o[nt]);
    }
  }

  // normalize + repack through LDS so global stores are contiguous b128s
  #pragma unroll
  for (int i = 0; i < 8; ++i) lrow[i] = 1.0f / lrow[i];
  _Float16* myO = &sOut[wave][0];
  #pragma unroll
  for (int nt = 0; nt < 8; ++nt)
    #pragma unroll
    for (int i = 0; i < 8; ++i)
      myO[(i + hi * 8) * OSTR + nt * 16 + mlane] = (_Float16)(o[nt][i] * lrow[i]);
  WAIT_DS0();
  {
    const int row = lane >> 1;           // 0..15
    const int cb  = (lane & 1) * 64;     // half-row of 64 halves
    const _Float16* src = myO + row * OSTR + cb;
    _Float16* dst = CTX + ((size_t)b * SEQ + qbase + row) * ATT + h * HDIM + cb;
    #pragma unroll
    for (int s8 = 0; s8 < 8; ++s8)
      gstore8(dst + s8 * 8, *(const v8h*)(src + s8 * 8));
  }
}

// ---------------- launch ----------------
extern "C" void kernel_launch(void* const* d_in, const int* in_sizes, int n_in,
                              void* d_out, int out_size, void* d_ws, size_t ws_size,
                              hipStream_t stream) {
  const float* query = (const float*)d_in[0];
  const float* Wq = (const float*)d_in[1];
  const float* bq = (const float*)d_in[2];
  const float* Wk = (const float*)d_in[3];
  const float* bk = (const float*)d_in[4];
  const float* Wv = (const float*)d_in[5];
  const float* bv = (const float*)d_in[6];
  const float* Wo = (const float*)d_in[7];
  const float* bo = (const float*)d_in[8];

  char* ws = (char*)d_ws;
  _Float16* XH  = (_Float16*)(ws + OFF_XH);
  _Float16* WQT = (_Float16*)(ws + OFF_WQT);
  _Float16* WKT = (_Float16*)(ws + OFF_WKT);
  _Float16* WVT = (_Float16*)(ws + OFF_WVT);
  _Float16* WOT = (_Float16*)(ws + OFF_WOT);
  _Float16* Qb  = (_Float16*)(ws + OFF_Q);
  _Float16* Kb  = (_Float16*)(ws + OFF_K);
  _Float16* VTb = (_Float16*)(ws + OFF_VT);
  _Float16* CTX = (_Float16*)(ws + OFF_CTX);

  // prep: casts + weight transposes
  cvt_f32_f16<<<(ROWS * DMODEL) / 256, 256, 0, stream>>>(query, XH, ROWS * DMODEL);
  transpose_w<<<(DMODEL * ATT) / 256, 256, 0, stream>>>(Wq, WQT, DMODEL, ATT);
  transpose_w<<<(DMODEL * ATT) / 256, 256, 0, stream>>>(Wk, WKT, DMODEL, ATT);
  transpose_w<<<(DMODEL * ATT) / 256, 256, 0, stream>>>(Wv, WVT, DMODEL, ATT);
  transpose_w<<<(ATT * OUTU) / 256, 256, 0, stream>>>(Wo, WOT, ATT, OUTU);

  // QKV projections (z = q,k,v)
  dim3 gq(ATT / 64, ROWS / 128, 3);
  gemm_qkv_kernel<<<gq, 256, 0, stream>>>(XH, WQT, WKT, WVT, bq, bk, bv, Qb, Kb, VTb);

  // causal flash attention
  dim3 ga(SEQ / 64, BATCH * HEADS);
  attn_kernel<<<ga, 128, 0, stream>>>(Qb, Kb, VTb, CTX);

  // output projection (f32 out)
  dim3 go(OUTU / 64, ROWS / 128);
  gemm_out_kernel<<<go, 256, 0, stream>>>(CTX, WOT, bo, (float*)d_out);
}
